// Net_53798760349958
// MI455X (gfx1250) — matrix operand order repaired and checked by reading, not compiled, not run on hardware
//
#include <hip/hip_runtime.h>

// ---------------------------------------------------------------------------
// Spiking MLP (16 -> 256 -> 256 -> 256 -> 16, T=16 IF steps) for gfx1250.
//
// Strategy (see analysis): layer-phased execution per block of 128 batch rows.
//  - Spikes are binary -> stored as 16-bit masks over T in LDS.
//  - Hidden-layer GEMMs run on v_wmma_f32_16x16x32_f16; the f32 WMMA
//    accumulator doubles as the IF membrane potential (soft reset).
//  - Weights live in LDS as f16 (128 KB), masks in two 64 KB LDS buffers.
// ---------------------------------------------------------------------------

typedef __attribute__((ext_vector_type(16))) _Float16 v16h;
typedef __attribute__((ext_vector_type(8)))  float    v8f;

#define THREADS        256
#define ROWS_PER_BLOCK 128
#define BATCH          32768
#define TSTEPS         16

typedef union { v16h h; unsigned u[8]; uint4 q[2]; } Frag16;

// Expand two ushort T-masks (packed in one dword) at time t into two f16
// values {0.0, 1.0} packed in one dword (0x3C00 == 1.0h).
__device__ __forceinline__ unsigned expand_pair(unsigned m, int t) {
  unsigned lo = (m >> t) & 1u;
  unsigned hi = (m >> (t + 16)) & 1u;
  return lo * 0x3C00u + hi * 0x3C000000u;
}

// Stage an f32 weight matrix from global into LDS as f16 (row-major kept).
__device__ __forceinline__ void stage_w_f16(_Float16* dst, const float* src,
                                            int nelem, int tid) {
  const float4* s = (const float4*)src;
  uint2* d = (uint2*)dst;
  for (int i = tid; i < (nelem >> 2); i += THREADS) {
    float4 v = s[i];
    union { uint2 u; _Float16 h[4]; } p;
    p.h[0] = (_Float16)v.x; p.h[1] = (_Float16)v.y;
    p.h[2] = (_Float16)v.z; p.h[3] = (_Float16)v.w;
    d[i] = p.u;
  }
}

// One 256->256 spiking FC layer over all T steps for this wave's 16 rows.
// mIn : [128][256] ushort LDS  (spike bitmask over T of previous layer)
// mOut: [128][256] ushort LDS  (spike bitmask over T of this layer)
// w   : [256 out][256 in] f16 LDS row-major
// WMMA accumulators acc[n] ARE the membrane potentials (v_reset=None).
__device__ __forceinline__ void if_fc_256(const unsigned short* mIn,
                                          unsigned short* mOut,
                                          const _Float16* w,
                                          int wave, int lane) {
  const int mRow = wave * 16;       // this wave's row base within the block
  const int l15  = lane & 15;
  const int lh   = lane >> 4;       // lane half (A/B K-group select)

  v8f      acc[16];                 // 16 N-tiles of 16x16 f32 state
  unsigned msk[16][4];              // spike bits, 2 ushort masks per dword
#pragma unroll
  for (int n = 0; n < 16; ++n) {
    acc[n] = (v8f){0.f,0.f,0.f,0.f,0.f,0.f,0.f,0.f};
    msk[n][0] = 0u; msk[n][1] = 0u; msk[n][2] = 0u; msk[n][3] = 0u;
  }

  for (int t = 0; t < TSTEPS; ++t) {
    for (int kc = 0; kc < 8; ++kc) {          // K in chunks of 32
      // --- A fragment: expand bit t of the input masks (A layout:
      //     lane holds row M=l15; VGPR0-3 K = 0..7 (+8 for hi lanes),
      //     VGPR4-7 K = 16..23 (+8 for hi lanes)) ---
      const uint4* pm =
          (const uint4*)(mIn + (mRow + l15) * 256 + kc * 32 + lh * 8);
      uint4 q0 = pm[0];
      uint4 q1 = pm[2];                       // +32 bytes => K+16
      Frag16 A;
      A.u[0] = expand_pair(q0.x, t); A.u[1] = expand_pair(q0.y, t);
      A.u[2] = expand_pair(q0.z, t); A.u[3] = expand_pair(q0.w, t);
      A.u[4] = expand_pair(q1.x, t); A.u[5] = expand_pair(q1.y, t);
      A.u[6] = expand_pair(q1.z, t); A.u[7] = expand_pair(q1.w, t);

#pragma unroll
      for (int n = 0; n < 16; ++n) {
        // --- B fragment: lane holds col N=l15 => row (n*16+l15) of w;
        //     K pairs contiguous => two ds_load_b128 ---
        const uint4* pw =
            (const uint4*)(w + (n * 16 + l15) * 256 + kc * 32 + lh * 8);
        Frag16 B;
        B.q[0] = pw[0];
        B.q[1] = pw[2];
        acc[n] = __builtin_amdgcn_wmma_f32_16x16x32_f16(
            false, A.h, false, B.h, (short)0, acc[n], false, false);
      }
    }
    // --- IF threshold + soft reset on the accumulators; record spike bits ---
#pragma unroll
    for (int n = 0; n < 16; ++n) {
#pragma unroll
      for (int j = 0; j < 8; ++j) {
        float v = acc[n][j];
        bool  s = v >= 1.0f;
        acc[n][j] = s ? (v - 1.0f) : v;
        msk[n][j >> 1] |= (s ? 1u : 0u) << (t + ((j & 1) << 4));
      }
    }
  }

  // --- write this layer's spike bitmask (D layout: lane l15 = col,
  //     VGPR j = row j + 8*lh) ---
#pragma unroll
  for (int n = 0; n < 16; ++n) {
#pragma unroll
    for (int j = 0; j < 8; ++j) {
      mOut[(mRow + j + 8 * lh) * 256 + n * 16 + l15] =
          (unsigned short)((msk[n][j >> 1] >> ((j & 1) * 16)) & 0xFFFFu);
    }
  }
}

__global__ __launch_bounds__(THREADS)
void snn_if_mlp_kernel(const float* __restrict__ x,
                       const float* __restrict__ w1,
                       const float* __restrict__ w2,
                       const float* __restrict__ w3,
                       const float* __restrict__ w4,
                       float* __restrict__ out) {
  __shared__ _Float16       sW[256 * 256];                 // 128 KB weights
  __shared__ unsigned short sM1[ROWS_PER_BLOCK * 256];     //  64 KB masks
  __shared__ unsigned short sM2[ROWS_PER_BLOCK * 256];     //  64 KB masks

  const int tid     = threadIdx.x;
  const int wave    = tid >> 5;      // wave32: 8 waves per block
  const int lane    = tid & 31;
  const int rowBase = blockIdx.x * ROWS_PER_BLOCK;

  // Stage w2 (f16 -> sW) and x / w1 (f32, overlaid into sM2 space).
  stage_w_f16(sW, w2, 256 * 256, tid);
  float* fbuf = (float*)sM2;  // [0,2048): x tile, [2048,6144): w1
  for (int i = tid; i < ROWS_PER_BLOCK * 16; i += THREADS)
    fbuf[i] = x[rowBase * 16 + i];
  for (int i = tid; i < 256 * 16; i += THREADS)
    fbuf[ROWS_PER_BLOCK * 16 + i] = w1[i];
  __syncthreads();

  // ---- Phase 1: a1 = x @ w1^T (K=16, scalar VALU) + elementwise IF scan.
  //      Input is identical across T, so layer-1 spikes come from a pure
  //      scan of a1; result stored as T-bitmask in sM1. ----
  {
    float wr[16];
#pragma unroll
    for (int k = 0; k < 16; ++k)
      wr[k] = fbuf[ROWS_PER_BLOCK * 16 + tid * 16 + k];
    for (int r = 0; r < ROWS_PER_BLOCK; ++r) {
      float a = 0.f;
#pragma unroll
      for (int k = 0; k < 16; ++k) a += fbuf[r * 16 + k] * wr[k];
      float v = 0.f;
      unsigned bits = 0u;
#pragma unroll
      for (int t = 0; t < TSTEPS; ++t) {
        v += a;
        if (v >= 1.0f) { v -= 1.0f; bits |= 1u << t; }
      }
      sM1[r * 256 + tid] = (unsigned short)bits;
    }
  }
  __syncthreads();

  // ---- Phase 2: layer 2 (w2), sM1 -> sM2 ----
  if_fc_256(sM1, sM2, sW, wave, lane);
  __syncthreads();
  stage_w_f16(sW, w3, 256 * 256, tid);
  __syncthreads();

  // ---- Phase 3: layer 3 (w3), sM2 -> sM1 ----
  if_fc_256(sM2, sM1, sW, wave, lane);
  __syncthreads();
  stage_w_f16(sW, w4, 16 * 256, tid);
  __syncthreads();

  // ---- Phase 4: layer 4 (256 -> 16), one N-tile per t, store to global ----
  {
    const int mRow = wave * 16;
    const int l15  = lane & 15;
    const int lh   = lane >> 4;
    for (int t = 0; t < TSTEPS; ++t) {
      v8f o = (v8f){0.f,0.f,0.f,0.f,0.f,0.f,0.f,0.f};
      for (int kc = 0; kc < 8; ++kc) {
        const uint4* pm =
            (const uint4*)(sM1 + (mRow + l15) * 256 + kc * 32 + lh * 8);
        uint4 q0 = pm[0];
        uint4 q1 = pm[2];
        Frag16 A;
        A.u[0] = expand_pair(q0.x, t); A.u[1] = expand_pair(q0.y, t);
        A.u[2] = expand_pair(q0.z, t); A.u[3] = expand_pair(q0.w, t);
        A.u[4] = expand_pair(q1.x, t); A.u[5] = expand_pair(q1.y, t);
        A.u[6] = expand_pair(q1.z, t); A.u[7] = expand_pair(q1.w, t);
        const uint4* pw = (const uint4*)(sW + l15 * 256 + kc * 32 + lh * 8);
        Frag16 B;
        B.q[0] = pw[0];
        B.q[1] = pw[2];
        o = __builtin_amdgcn_wmma_f32_16x16x32_f16(
            false, A.h, false, B.h, (short)0, o, false, false);
      }
#pragma unroll
      for (int j = 0; j < 8; ++j) {
        out[(size_t)t * (BATCH * 16) +
            (size_t)(rowBase + mRow + j + 8 * lh) * 16 + l15] = o[j];
      }
    }
  }
}

extern "C" void kernel_launch(void* const* d_in, const int* in_sizes, int n_in,
                              void* d_out, int out_size, void* d_ws,
                              size_t ws_size, hipStream_t stream) {
  (void)in_sizes; (void)n_in; (void)out_size; (void)d_ws; (void)ws_size;
  const float* x  = (const float*)d_in[0];
  const float* w1 = (const float*)d_in[1];
  const float* w2 = (const float*)d_in[2];
  const float* w3 = (const float*)d_in[3];
  const float* w4 = (const float*)d_in[4];
  float* out = (float*)d_out;

  dim3 grid(BATCH / ROWS_PER_BLOCK);   // 256 blocks
  dim3 block(THREADS);                 // 8 wave32 waves
  snn_if_mlp_kernel<<<grid, block, 0, stream>>>(x, w1, w2, w3, w4, out);
}
